// MultiHeadHighLevelAllocator_14912126451721
// MI455X (gfx1250) — matrix operand order repaired and checked by reading, not compiled, not run on hardware
//
#include <hip/hip_runtime.h>
#include <hip/hip_bf16.h>
#include <stdint.h>

#define B_    16
#define U_    64
#define T_    128
#define E_    128
#define H_    4
#define HID_  256
#define ENCH_ 128
#define IND_  32

typedef __attribute__((ext_vector_type(16))) __bf16 bf16x16;
typedef __attribute__((ext_vector_type(8)))  float  f32x8;

__device__ __forceinline__ unsigned short f2bf(float f) {
    unsigned int x = __float_as_uint(f);
    unsigned int r = ((x >> 16) & 1u) + 0x7FFFu;   // round-to-nearest-even
    return (unsigned short)((x + r) >> 16);
}

// ---------------- prep kernels ----------------
__global__ void cvt_bf16(const float* __restrict__ src, unsigned short* __restrict__ dst, int n) {
    int i = blockIdx.x * blockDim.x + threadIdx.x;
    if (i < n) dst[i] = f2bf(src[i]);
}

// src is (K,N) f32 row-major; dst is (N,K) bf16 row-major (i.e. transposed)
__global__ void transpose_bf16(const float* __restrict__ src, unsigned short* __restrict__ dst,
                               int K, int N) {
    int i = blockIdx.x * blockDim.x + threadIdx.x;
    if (i < K * N) {
        int n = i / K, k = i - n * K;
        dst[n * K + k] = f2bf(src[k * N + n]);
    }
}

// hp[h,d] = fb1[d] + sum_e head_q[h,e] * fw1[e,d]   (fw1 top half, fp32)
__global__ void head_proj(const float* __restrict__ head_q, const float* __restrict__ fw1,
                          const float* __restrict__ fb1, float* __restrict__ hp) {
    int i = blockIdx.x * blockDim.x + threadIdx.x;
    if (i >= H_ * HID_) return;
    int h = i / HID_, d = i - h * HID_;
    float acc = fb1[d];
    for (int e = 0; e < E_; ++e)
        acc = fmaf(head_q[h * E_ + e], fw1[e * HID_ + d], acc);
    hp[i] = acc;
}

// ---------------- WMMA GEMM: C[M,N] = act(A[M,K] @ B[K,N] + bias) ----------------
// A: bf16 row-major (M,K); BT: bf16 (N,K) [= B transposed]; bias: f32[N] or null.
// 128 threads = 4 waves; wave w owns the 16x16 tile at (blockIdx.y*16, blockIdx.x*64 + w*16).
__global__ __launch_bounds__(128) void gemm_wmma_bf16(
    const unsigned short* __restrict__ A,
    const unsigned short* __restrict__ BT,
    const float* __restrict__ bias,
    void* __restrict__ Cout,
    int M, int N, int K, int relu, int out_bf16)
{
    const int lane  = threadIdx.x & 31;
    const int wave  = threadIdx.x >> 5;
    const int mBase = blockIdx.y * 16;
    const int nBase = blockIdx.x * 64 + wave * 16;
    const int row   = mBase + (lane & 15);
    const int col   = nBase + (lane & 15);
    const int aK    = (lane >> 4) * 8;    // A: K-half select per ISA layout
    const int bK    = (lane >> 4) * 8;    // B: K-half select per ISA layout

    f32x8 acc = {0.f, 0.f, 0.f, 0.f, 0.f, 0.f, 0.f, 0.f};
    for (int ko = 0; ko < K; ko += 32) {
        union { bf16x16 v; uint4 q[2]; } a, b;
        const unsigned short* ap = A  + (size_t)row * K + ko + aK;
        const unsigned short* bp = BT + (size_t)col * K + ko + bK;
        a.q[0] = *(const uint4*)(ap);        // K = aK .. aK+7
        a.q[1] = *(const uint4*)(ap + 16);   // K = aK+16 .. aK+23
        b.q[0] = *(const uint4*)(bp);        // K = bK .. bK+7
        b.q[1] = *(const uint4*)(bp + 16);   // K = bK+16 .. bK+23
        acc = __builtin_amdgcn_wmma_f32_16x16x32_bf16(
                  false, a.v, false, b.v, (short)0, acc, false, false);
    }

    const float bval  = bias ? bias[col] : 0.f;
    const int   rBase = (lane >> 4) * 8;
    if (out_bf16) {
        unsigned short* C = (unsigned short*)Cout;
        #pragma unroll
        for (int i = 0; i < 8; ++i) {
            float v = acc[i] + bval;
            if (relu) v = fmaxf(v, 0.f);
            C[(size_t)(mBase + rBase + i) * N + col] = f2bf(v);
        }
    } else {
        float* C = (float*)Cout;
        #pragma unroll
        for (int i = 0; i < 8; ++i) {
            float v = acc[i] + bval;
            if (relu) v = fmaxf(v, 0.f);
            C[(size_t)(mBase + rBase + i) * N + col] = v;
        }
    }
}

// ---------------- scoring stage ----------------
// out[b,h,u,t] = fb2 + sum_d relu(ueh[b,u,d] + hp[h,d] + kh[b,t,d]) * fw2[d]
// Block: 256 threads, covers 16 u x 128 t for one (b,h). d tiled by 64 through LDS.
#define DC 64
__global__ __launch_bounds__(256) void fuse_score(
    const float* __restrict__ ueh,   // (B*U, HID)
    const float* __restrict__ kh,    // (B*T, HID)
    const float* __restrict__ hp,    // (H, HID)
    const float* __restrict__ fw2,   // (HID,)
    const float* __restrict__ fb2,   // (1,)
    float* __restrict__ out)         // (B,H,U,T)
{
    __shared__ float ks[T_ * (DC + 1)];   // padded: stride 65 -> conflict-free
    __shared__ float qs[16 * (DC + 1)];
    __shared__ float ws[DC];

    const int tid = threadIdx.x;
    const int bh  = blockIdx.y;
    const int b   = bh >> 2, h = bh & 3;
    const int u0  = blockIdx.x * 16;
    const int t   = tid & 127;
    const int ug  = (tid >> 7) * 8;

    float acc[8];
    #pragma unroll
    for (int i = 0; i < 8; ++i) acc[i] = 0.f;

    for (int dc = 0; dc < HID_; dc += DC) {
        // stage k tile (128 x 64), coalesced
        #pragma unroll
        for (int r = 0; r < (T_ * DC) / 256; ++r) {
            int idx = r * 256 + tid;
            int tt = idx >> 6, dd = idx & 63;
            ks[tt * (DC + 1) + dd] = kh[((size_t)b * T_ + tt) * HID_ + dc + dd];
        }
        // stage q tile (16 x 64) with head bias folded in
        #pragma unroll
        for (int r = 0; r < (16 * DC) / 256; ++r) {
            int idx = r * 256 + tid;
            int uu = idx >> 6, dd = idx & 63;
            qs[uu * (DC + 1) + dd] =
                ueh[((size_t)b * U_ + u0 + uu) * HID_ + dc + dd] + hp[h * HID_ + dc + dd];
        }
        if (tid < DC) ws[tid] = fw2[dc + tid];
        __syncthreads();

        #pragma unroll 4
        for (int d = 0; d < DC; ++d) {
            float kv = ks[t * (DC + 1) + d];
            float wv = ws[d];
            #pragma unroll
            for (int i = 0; i < 8; ++i) {
                float x = qs[(ug + i) * (DC + 1) + d] + kv;
                acc[i] = fmaf(fmaxf(x, 0.f), wv, acc[i]);
            }
        }
        __syncthreads();
    }

    const float bb = fb2[0];
    #pragma unroll
    for (int i = 0; i < 8; ++i)
        out[(((size_t)b * H_ + h) * U_ + (u0 + ug + i)) * T_ + t] = acc[i] + bb;
}

// ---------------- host launch ----------------
extern "C" void kernel_launch(void* const* d_in, const int* in_sizes, int n_in,
                              void* d_out, int out_size, void* d_ws, size_t ws_size,
                              hipStream_t stream) {
    const float* uav_feat = (const float*)d_in[0];
    const float* task_feat= (const float*)d_in[1];
    const float* uw0 = (const float*)d_in[2];  const float* ub0 = (const float*)d_in[3];
    const float* uw1 = (const float*)d_in[4];  const float* ub1 = (const float*)d_in[5];
    const float* uw2 = (const float*)d_in[6];  const float* ub2 = (const float*)d_in[7];
    const float* tw0 = (const float*)d_in[8];  const float* tb0 = (const float*)d_in[9];
    const float* tw1 = (const float*)d_in[10]; const float* tb1 = (const float*)d_in[11];
    const float* tw2 = (const float*)d_in[12]; const float* tb2 = (const float*)d_in[13];
    const float* head_q = (const float*)d_in[14];
    const float* fw1 = (const float*)d_in[15]; const float* fb1 = (const float*)d_in[16];
    const float* fw2 = (const float*)d_in[17]; const float* fb2 = (const float*)d_in[18];
    float* out = (float*)d_out;

    // carve workspace
    char* base = (char*)d_ws;
    size_t off = 0;
    auto carve = [&](size_t bytes) -> char* {
        char* p = base + off;
        off = (off + bytes + 255) & ~(size_t)255;
        return p;
    };
    unsigned short* uw0T  = (unsigned short*)carve(128u * 32  * 2);
    unsigned short* uw1T  = (unsigned short*)carve(128u * 128 * 2);
    unsigned short* uw2T  = (unsigned short*)carve(128u * 128 * 2);
    unsigned short* tw0T  = (unsigned short*)carve(128u * 32  * 2);
    unsigned short* tw1T  = (unsigned short*)carve(128u * 128 * 2);
    unsigned short* tw2T  = (unsigned short*)carve(128u * 128 * 2);
    unsigned short* fw1aT = (unsigned short*)carve(256u * 128 * 2);
    unsigned short* fw1bT = (unsigned short*)carve(256u * 128 * 2);
    unsigned short* xu    = (unsigned short*)carve(1024u * 32  * 2);
    unsigned short* xt    = (unsigned short*)carve(2048u * 32  * 2);
    unsigned short* hu0   = (unsigned short*)carve(1024u * 128 * 2);
    unsigned short* hu1   = (unsigned short*)carve(1024u * 128 * 2);
    unsigned short* ue    = (unsigned short*)carve(1024u * 128 * 2);
    unsigned short* ht0   = (unsigned short*)carve(2048u * 128 * 2);
    unsigned short* ht1   = (unsigned short*)carve(2048u * 128 * 2);
    unsigned short* te    = (unsigned short*)carve(2048u * 128 * 2);
    float* ueh = (float*)carve(1024u * 256 * 4);
    float* kh  = (float*)carve(2048u * 256 * 4);
    float* hp  = (float*)carve(4u   * 256 * 4);
    (void)ws_size; (void)n_in; (void)in_sizes; (void)out_size;

    // --- prep: activations to bf16, weights to transposed bf16 ---
    cvt_bf16<<<(1024*32 + 255)/256, 256, 0, stream>>>(uav_feat,  xu, 1024*32);
    cvt_bf16<<<(2048*32 + 255)/256, 256, 0, stream>>>(task_feat, xt, 2048*32);
    transpose_bf16<<<(32*128  + 255)/256, 256, 0, stream>>>(uw0, uw0T, 32, 128);
    transpose_bf16<<<(128*128 + 255)/256, 256, 0, stream>>>(uw1, uw1T, 128, 128);
    transpose_bf16<<<(128*128 + 255)/256, 256, 0, stream>>>(uw2, uw2T, 128, 128);
    transpose_bf16<<<(32*128  + 255)/256, 256, 0, stream>>>(tw0, tw0T, 32, 128);
    transpose_bf16<<<(128*128 + 255)/256, 256, 0, stream>>>(tw1, tw1T, 128, 128);
    transpose_bf16<<<(128*128 + 255)/256, 256, 0, stream>>>(tw2, tw2T, 128, 128);
    transpose_bf16<<<(128*256 + 255)/256, 256, 0, stream>>>(fw1,             fw1aT, 128, 256);
    transpose_bf16<<<(128*256 + 255)/256, 256, 0, stream>>>(fw1 + 128 * 256, fw1bT, 128, 256);
    head_proj<<<(H_*HID_ + 255)/256, 256, 0, stream>>>(head_q, fw1, fb1, hp);

    // --- UAV encoder chain (WMMA) ---
    gemm_wmma_bf16<<<dim3(2, 64),  128, 0, stream>>>(xu,  uw0T, ub0, hu0, 1024, 128, 32,  1, 1);
    gemm_wmma_bf16<<<dim3(2, 64),  128, 0, stream>>>(hu0, uw1T, ub1, hu1, 1024, 128, 128, 1, 1);
    gemm_wmma_bf16<<<dim3(2, 64),  128, 0, stream>>>(hu1, uw2T, ub2, ue,  1024, 128, 128, 0, 1);
    gemm_wmma_bf16<<<dim3(4, 64),  128, 0, stream>>>(ue,  fw1aT, (const float*)nullptr,
                                                     ueh, 1024, 256, 128, 0, 0);
    // --- Task encoder chain (WMMA) ---
    gemm_wmma_bf16<<<dim3(2, 128), 128, 0, stream>>>(xt,  tw0T, tb0, ht0, 2048, 128, 32,  1, 1);
    gemm_wmma_bf16<<<dim3(2, 128), 128, 0, stream>>>(ht0, tw1T, tb1, ht1, 2048, 128, 128, 1, 1);
    gemm_wmma_bf16<<<dim3(2, 128), 128, 0, stream>>>(ht1, tw2T, tb2, te,  2048, 128, 128, 0, 1);
    gemm_wmma_bf16<<<dim3(4, 128), 128, 0, stream>>>(te,  fw1bT, (const float*)nullptr,
                                                     kh,  2048, 256, 128, 0, 0);

    // --- fused additive-attention scoring ---
    fuse_score<<<dim3(U_/16, B_*H_), 256, 0, stream>>>(ueh, kh, hp, fw2, fb2, out);
}
